// pool_corner_71373766525607
// MI455X (gfx1250) — matrix operand order repaired and checked by reading, not compiled, not run on hardware
//
#include <hip/hip_runtime.h>

typedef _Float16 v16h __attribute__((ext_vector_type(16)));
typedef _Float16 v8h  __attribute__((ext_vector_type(8)));
typedef float    v8f  __attribute__((ext_vector_type(8)));
typedef __attribute__((address_space(3))) _Float16 lds_f16;

#define HH 128
#define WW 128
#define NB 8

__device__ __forceinline__ unsigned lds_off(const _Float16* p) {
    return (unsigned)(unsigned long long)(lds_f16*)p;
}

// ---------------------------------------------------------------------------
// NCHW f32 -> NHWC f16 transpose-convert (for x only). LDS tile 32c x 64w.
// ---------------------------------------------------------------------------
__global__ __launch_bounds__(256)
void cvt_nchw_to_nhwc(const float* __restrict__ x, _Float16* __restrict__ xh)
{
    __shared__ float s[32 * 65];
    const int tid = threadIdx.x;
    const int w0  = blockIdx.x * 64;
    const int c0  = blockIdx.y * 32;
    const int h   = blockIdx.z & (HH - 1);
    const int b   = blockIdx.z >> 7;
    for (int i = tid; i < 32 * 64; i += 256) {
        const int c = i >> 6, w = i & 63;
        s[c * 65 + w] = x[(((size_t)b * 256 + (c0 + c)) * HH + h) * WW + (w0 + w)];
    }
    __syncthreads();
    for (int i = tid; i < 32 * 64; i += 256) {
        const int w = i >> 5, c = i & 31;
        xh[(((size_t)b * HH + h) * WW + (w0 + w)) * 256 + (c0 + c)] =
            (_Float16)s[c * 65 + w];
    }
}

// ---------------------------------------------------------------------------
// Pre-pack conv weights: OIHW f32 -> f16 in WMMA-fragment-ready LDS layout
// out[((chunk*TAPS + tap)*COUT + co)*32 + ci] = w[co][chunk*32+ci][tap]
// ---------------------------------------------------------------------------
__global__ __launch_bounds__(256)
void pack_weights(const float* __restrict__ wgt, _Float16* __restrict__ out,
                  int CIN, int COUT, int TAPS)
{
    const size_t total = (size_t)CIN * COUT * TAPS;
    for (size_t i = (size_t)blockIdx.x * 256 + threadIdx.x; i < total;
         i += (size_t)gridDim.x * 256) {
        const int ci    = (int)(i & 31);
        const size_t r  = i >> 5;
        const int co    = (int)(r % COUT);
        const size_t q  = r / COUT;
        const int tap   = (int)(q % TAPS);
        const int chunk = (int)(q / TAPS);
        const int cig   = chunk * 32 + ci;
        out[i] = (_Float16)wgt[((size_t)co * CIN + cig) * TAPS + tap];
    }
}

// ---------------------------------------------------------------------------
// Implicit-GEMM conv (KSxKS, SAME) on NHWC f16 activations, f16 pre-packed
// weights. Grid: (H, Cout/64, B). Block: 256 threads = 8 waves.
// Block tile: 64 couts x 128 pixels; each wave = 16 pixels x 64 couts
// (4 accumulators, one B fragment reused across 4 A fragments per tap).
// Double-buffered dynamic-LDS pipeline: chunk k+1 is copied with
// global_load_async_to_lds_b128 (ASYNCcnt) while chunk k's WMMAs execute;
// one s_wait_asynccnt + one barrier per chunk.
// LDS layout (halfs): [sX buf0][sX buf1][sW buf0][sW buf1][BN scale|shift]
// ---------------------------------------------------------------------------
template<int CIN, int KS, bool BN, bool RELUO, bool SUM2, bool RELUI, bool OUT32>
__global__ __launch_bounds__(256)
void conv_wmma_kernel(const _Float16* __restrict__ in0,
                      const _Float16* __restrict__ in1,
                      const _Float16* __restrict__ wpk,
                      const float* __restrict__ bn_g,
                      const float* __restrict__ bn_b,
                      const float* __restrict__ bn_m,
                      const float* __restrict__ bn_v,
                      void* __restrict__ out_raw,
                      int cout_total)
{
    constexpr int PAD  = KS / 2;
    constexpr int WT   = WW + (KS - 1);
    constexpr int TAPS = KS * KS;
    constexpr int CCH  = 32;               // K per WMMA
    constexpr int COT  = 64;               // couts per block
    constexpr int XSZ  = KS * WT * CCH;    // halfs per activation buffer
    constexpr int WSZ  = TAPS * COT * CCH; // halfs per weight buffer
    constexpr bool AX  = !(SUM2 || RELUI); // async-copy activations?

    extern __shared__ _Float16 smem[];     // 2*XSZ + 2*WSZ halfs + BN floats
    const unsigned sbase = lds_off(smem);  // LDS byte offset of smem

    const int h    = blockIdx.x;
    const int co0  = blockIdx.y * COT;
    const int bimg = blockIdx.z;
    const int tid  = threadIdx.x;
    const int lane = tid & 31;
    const int wv   = tid >> 5;
    const int n    = lane & 15;   // N (pixel) / M (cout) lane index
    const int hsel = lane >> 4;   // K-half selector (16-bit WMMA layout)
    const int wb   = wv * 16;

    float* s_scale = (float*)(smem + 2 * XSZ + 2 * WSZ);
    float* s_shift = s_scale + COT;
    if constexpr (BN) {
        if (tid < COT) {
            const int co = co0 + tid;
            const float sc = bn_g[co] * rsqrtf(bn_v[co] + 1e-5f);
            s_scale[tid] = sc;
            s_shift[tid] = bn_b[co] - bn_m[co] * sc;
        }
    }

    const unsigned long long in0b =
        (unsigned long long)(const void*)(in0 + (size_t)bimg * HH * WW * CIN);
    const unsigned long long wpkb = (unsigned long long)(const void*)wpk;

    // ---- prologue: zero halo slots of both buffers (mask is chunk-invariant)
    #pragma unroll
    for (int p = 0; p < 2; ++p) {
        _Float16* bx = smem + p * XSZ;
        for (int s = tid; s < KS * WT * 4; s += 256) {
            const int pos = s >> 2;
            const int seg = s & 3;
            const int w   = pos % WT;
            const int row = pos / WT;
            const int sw  = w - PAD;
            const int sh  = h + row - PAD;
            if (!((unsigned)sw < (unsigned)WW && (unsigned)sh < (unsigned)HH))
                *reinterpret_cast<v8h*>(&bx[pos * CCH + seg * 8]) = (v8h){};
        }
    }

    // ---- stage one chunk into buffer p (async where possible) ----
    auto issue = [&](int ci0, int p) {
        _Float16* bx = smem + p * XSZ;
        const unsigned bxo = sbase + (unsigned)(p * XSZ) * 2u;
        for (int s = tid; s < KS * WT * 4; s += 256) {
            const int pos = s >> 2;
            const int seg = s & 3;
            const int w   = pos % WT;
            const int row = pos / WT;
            const int sw  = w - PAD;
            const int sh  = h + row - PAD;
            const bool ok = (unsigned)sw < (unsigned)WW && (unsigned)sh < (unsigned)HH;
            if (!ok) continue;  // halo already zeroed in prologue
            if constexpr (AX) {
                const unsigned ldsa = bxo + (unsigned)(pos * CCH + seg * 8) * 2u;
                const unsigned goff =
                    (unsigned)(((sh * WW + sw) * CIN + ci0 + seg * 8) * 2);
                asm volatile("global_load_async_to_lds_b128 %0, %1, %2 offset:0"
                             :: "v"(ldsa), "v"(goff), "s"(in0b) : "memory");
            } else {
                const size_t gi =
                    (((size_t)bimg * HH + sh) * WW + sw) * CIN + ci0 + seg * 8;
                v8h v = *reinterpret_cast<const v8h*>(in0 + gi);
                if constexpr (SUM2)
                    v += *reinterpret_cast<const v8h*>(in1 + gi);
                if constexpr (RELUI) {
                    #pragma unroll
                    for (int e = 0; e < 8; ++e)
                        v[e] = v[e] > (_Float16)0.0f ? v[e] : (_Float16)0.0f;
                }
                *reinterpret_cast<v8h*>(&bx[pos * CCH + seg * 8]) = v;
            }
        }
        // pre-packed weights: pure async 16B copies
        const unsigned bwo = sbase + (unsigned)(2 * XSZ + p * WSZ) * 2u;
        const int chunk = ci0 / CCH;
        for (int s = tid; s < TAPS * 256; s += 256) {
            const int tap = s >> 8;
            const int u   = s & 255;
            const unsigned ldsa = bwo + (unsigned)(tap * 4096 + u * 16);
            const unsigned goff = (unsigned)(
                (((chunk * TAPS + tap) * cout_total + co0) * CCH) * 2 + u * 16);
            asm volatile("global_load_async_to_lds_b128 %0, %1, %2 offset:0"
                         :: "v"(ldsa), "v"(goff), "s"(wpkb) : "memory");
        }
    };

    v8f acc[4] = {};
    constexpr int NCH = CIN / CCH;

    issue(0, 0);
    for (int k = 0; k < NCH; ++k) {
        const int p = k & 1;
        asm volatile("s_wait_asynccnt 0" ::: "memory");  // chunk k copies done
        __syncthreads();           // all waves done with buffer 1-p (chunk k-1)
        if (k + 1 < NCH) issue((k + 1) * CCH, 1 - p);    // overlaps with WMMAs

        const _Float16* bx = smem + p * XSZ;
        const _Float16* bw = smem + 2 * XSZ + p * WSZ;
        #pragma unroll
        for (int tap = 0; tap < TAPS; ++tap) {
            const int kh = tap / KS;
            const int kw = tap % KS;
            const int wpos = wb + n + kw;
            const v16h bfrag = *reinterpret_cast<const v16h*>(
                &bx[(kh * WT + wpos) * CCH + hsel * 16]);
            #pragma unroll
            for (int t = 0; t < 4; ++t) {
                const _Float16* ap = &bw[(tap * COT + t * 16 + n) * CCH];
                const v8h alo = *reinterpret_cast<const v8h*>(ap + hsel * 16);
                const v8h ahi = *reinterpret_cast<const v8h*>(ap + hsel * 16 + 8);
                const v16h a = __builtin_shufflevector(
                    alo, ahi, 0,1,2,3,4,5,6,7,8,9,10,11,12,13,14,15);
                acc[t] = __builtin_amdgcn_wmma_f32_16x16x32_f16(
                    false, a, false, bfrag, (short)0, acc[t], false, false);
            }
        }
    }

    // ---- epilogue: lane n = pixel, VGPR r of tile t -> co = t*16+hsel*8+r --
    #pragma unroll
    for (int t = 0; t < 4; ++t) {
        if constexpr (OUT32) {
            float* out = (float*)out_raw;  // NCHW f32 (final output)
            #pragma unroll
            for (int r = 0; r < 8; ++r) {
                const int col = t * 16 + hsel * 8 + r;
                float val = acc[t][r];
                if constexpr (BN)    val = val * s_scale[col] + s_shift[col];
                if constexpr (RELUO) val = fmaxf(val, 0.0f);
                out[(((size_t)bimg * cout_total + (co0 + col)) * HH + h) * WW +
                    (wb + n)] = val;
            }
        } else {
            _Float16* out = (_Float16*)out_raw;  // NHWC f16 (intermediate)
            v8h o;
            #pragma unroll
            for (int r = 0; r < 8; ++r) {
                const int col = t * 16 + hsel * 8 + r;
                float val = acc[t][r];
                if constexpr (BN)    val = val * s_scale[col] + s_shift[col];
                if constexpr (RELUO) val = fmaxf(val, 0.0f);
                o[r] = (_Float16)val;
            }
            const size_t go = (((size_t)bimg * HH + h) * WW + (wb + n)) *
                                  cout_total + co0 + t * 16 + hsel * 8;
            *reinterpret_cast<v8h*>(out + go) = o;  // 16B contiguous store
        }
    }
}

// ---------------------------------------------------------------------------
// TopPool on NHWC f16: reverse cummax along H, in place. Thread per (b,w,c).
// ---------------------------------------------------------------------------
__global__ __launch_bounds__(256)
void top_pool_nhwc(_Float16* __restrict__ t)
{
    const int i = blockIdx.x * 256 + threadIdx.x;   // over NB*W*128
    const int c = i & 127;
    const int w = (i >> 7) & (WW - 1);
    const int b = i >> 14;
    _Float16* p = t + (((size_t)b * HH) * WW + w) * 128 + c;
    const size_t st = (size_t)WW * 128;
    _Float16 cur = p[(HH - 1) * st];
    for (int hh = HH - 2; hh >= 0; --hh) {
        const _Float16 v = p[hh * st];
        cur = v > cur ? v : cur;
        p[hh * st] = cur;
    }
}

// ---------------------------------------------------------------------------
// LeftPool on NHWC f16: reverse cummax along W, in place. Thread per (b,h,c).
// ---------------------------------------------------------------------------
__global__ __launch_bounds__(256)
void left_pool_nhwc(_Float16* __restrict__ t)
{
    const int i = blockIdx.x * 256 + threadIdx.x;   // over NB*H*128
    const int c = i & 127;
    const int hh = (i >> 7) & (HH - 1);
    const int b = i >> 14;
    _Float16* p = t + (((size_t)b * HH + hh) * WW) * 128 + c;
    _Float16 cur = p[(WW - 1) * 128];
    for (int w = WW - 2; w >= 0; --w) {
        const _Float16 v = p[w * 128];
        cur = v > cur ? v : cur;
        p[w * 128] = cur;
    }
}

// ---------------------------------------------------------------------------
extern "C" void kernel_launch(void* const* d_in, const int* in_sizes, int n_in,
                              void* d_out, int out_size, void* d_ws, size_t ws_size,
                              hipStream_t stream)
{
    (void)in_sizes; (void)n_in; (void)out_size; (void)ws_size;

    const float* x       = (const float*)d_in[0];
    const float* lc1_w   = (const float*)d_in[1];
    const float* lc1_g   = (const float*)d_in[2];
    const float* lc1_b   = (const float*)d_in[3];
    const float* lc1_m   = (const float*)d_in[4];
    const float* lc1_v   = (const float*)d_in[5];
    const float* p1c1_w  = (const float*)d_in[6];
    const float* p1c1_g  = (const float*)d_in[7];
    const float* p1c1_b  = (const float*)d_in[8];
    const float* p1c1_m  = (const float*)d_in[9];
    const float* p1c1_v  = (const float*)d_in[10];
    const float* lc2_w   = (const float*)d_in[11];
    const float* lc2_g   = (const float*)d_in[12];
    const float* lc2_b   = (const float*)d_in[13];
    const float* lc2_m   = (const float*)d_in[14];
    const float* lc2_v   = (const float*)d_in[15];
    const float* p2c1_w  = (const float*)d_in[16];
    const float* p2c1_g  = (const float*)d_in[17];
    const float* p2c1_b  = (const float*)d_in[18];
    const float* p2c1_m  = (const float*)d_in[19];
    const float* p2c1_v  = (const float*)d_in[20];
    const float* p1lc_w  = (const float*)d_in[21];
    const float* p2lc_w  = (const float*)d_in[22];
    const float* pconv1_w = (const float*)d_in[23];
    const float* pbn1_g  = (const float*)d_in[24];
    const float* pbn1_b  = (const float*)d_in[25];
    const float* pbn1_m  = (const float*)d_in[26];
    const float* pbn1_v  = (const float*)d_in[27];
    const float* conv1_w = (const float*)d_in[28];
    const float* bn1_g   = (const float*)d_in[29];
    const float* bn1_b   = (const float*)d_in[30];
    const float* bn1_m   = (const float*)d_in[31];
    const float* bn1_v   = (const float*)d_in[32];
    const float* c2_w    = (const float*)d_in[33];
    const float* c2_g    = (const float*)d_in[34];
    const float* c2_b    = (const float*)d_in[35];
    const float* c2_m    = (const float*)d_in[36];
    const float* c2_v    = (const float*)d_in[37];

    _Float16* wsh = (_Float16*)d_ws;
    const size_t SC = (size_t)NB * HH * WW;     // pixels = 131072
    _Float16* xh = wsh;                          // 256-ch f16 NHWC
    _Float16* A  = wsh + SC * 256;               // 128-ch scratch
    _Float16* Bf = A + SC * 128;
    _Float16* Cf = Bf + SC * 128;
    _Float16* Df = Cf + SC * 128;
    _Float16* E  = A;                            // 256-ch, aliases A+B
    _Float16* F  = Cf;                           // 256-ch, aliases C+D
    _Float16* wp = Df + SC * 128;                // packed weight area (f16)
    _Float16* lc1_p   = wp;                wp += (size_t)256 * 128 * 9;
    _Float16* p1c1_p  = wp;                wp += (size_t)256 * 128 * 9;
    _Float16* lc2_p   = wp;                wp += (size_t)256 * 128 * 9;
    _Float16* p2c1_p  = wp;                wp += (size_t)256 * 128 * 9;
    _Float16* p1lc_p  = wp;                wp += (size_t)128 * 128 * 9;
    _Float16* p2lc_p  = wp;                wp += (size_t)128 * 128 * 9;
    _Float16* pconv1_p = wp;               wp += (size_t)128 * 256 * 9;
    _Float16* conv1_p = wp;                wp += (size_t)256 * 256 * 1;
    _Float16* c2_p    = wp;                wp += (size_t)256 * 256 * 9;
    float* outp  = (float*)d_out;

    const dim3 g128(HH, 128 / 64, NB);
    const dim3 g256(HH, 256 / 64, NB);
    const int pool_blocks = (int)(SC * 128 / (size_t)(HH * 256));

    // dynamic LDS: 2*(sX + sW) + BN arrays
    const int SH3 = 2 * (3 * 130 * 32 + 9 * 64 * 32) * 2 + 64 * 8;  // 124160 B
    const int SH1 = 2 * (1 * 128 * 32 + 1 * 64 * 32) * 2 + 64 * 8;  //  25088 B

    auto KA = conv_wmma_kernel<256, 3, true,  true,  false, false, false>;
    auto KB = conv_wmma_kernel<128, 3, false, false, true,  false, false>;
    auto KC = conv_wmma_kernel<128, 3, true,  false, true,  false, false>;
    auto KD = conv_wmma_kernel<256, 1, true,  false, false, false, false>;
    auto KE = conv_wmma_kernel<256, 3, true,  true,  true,  true,  true>;
    (void)hipFuncSetAttribute((const void*)KA,
        hipFuncAttributeMaxDynamicSharedMemorySize, SH3);
    (void)hipFuncSetAttribute((const void*)KB,
        hipFuncAttributeMaxDynamicSharedMemorySize, SH3);
    (void)hipFuncSetAttribute((const void*)KC,
        hipFuncAttributeMaxDynamicSharedMemorySize, SH3);
    (void)hipFuncSetAttribute((const void*)KD,
        hipFuncAttributeMaxDynamicSharedMemorySize, SH1);
    (void)hipFuncSetAttribute((const void*)KE,
        hipFuncAttributeMaxDynamicSharedMemorySize, SH3);

    // one-time prep: x transpose-convert + weight packing
    cvt_nchw_to_nhwc<<<dim3(WW / 64, 256 / 32, NB * HH), 256, 0, stream>>>(x, xh);
    pack_weights<<<1152, 256, 0, stream>>>(lc1_w,   lc1_p,   256, 128, 9);
    pack_weights<<<1152, 256, 0, stream>>>(p1c1_w,  p1c1_p,  256, 128, 9);
    pack_weights<<<1152, 256, 0, stream>>>(lc2_w,   lc2_p,   256, 128, 9);
    pack_weights<<<1152, 256, 0, stream>>>(p2c1_w,  p2c1_p,  256, 128, 9);
    pack_weights<<<576,  256, 0, stream>>>(p1lc_w,  p1lc_p,  128, 128, 9);
    pack_weights<<<576,  256, 0, stream>>>(p2lc_w,  p2lc_p,  128, 128, 9);
    pack_weights<<<1152, 256, 0, stream>>>(pconv1_w, pconv1_p, 128, 256, 9);
    pack_weights<<<256,  256, 0, stream>>>(conv1_w, conv1_p, 256, 256, 1);
    pack_weights<<<2304, 256, 0, stream>>>(c2_w,    c2_p,    256, 256, 9);

    // branch 1
    KA<<<g128, 256, SH3, stream>>>(xh, nullptr, lc1_p,
        lc1_g, lc1_b, lc1_m, lc1_v, A, 128);                              // look_conv1
    KA<<<g128, 256, SH3, stream>>>(xh, nullptr, p1c1_p,
        p1c1_g, p1c1_b, p1c1_m, p1c1_v, Bf, 128);                         // p1_conv1
    left_pool_nhwc<<<pool_blocks, 256, 0, stream>>>(A);                   // look2
    KB<<<g128, 256, SH3, stream>>>(Bf, A, p1lc_p,
        nullptr, nullptr, nullptr, nullptr, Cf, 128);                     // p1_look
    top_pool_nhwc<<<pool_blocks, 256, 0, stream>>>(Cf);                   // pool1

    // branch 2
    KA<<<g128, 256, SH3, stream>>>(xh, nullptr, lc2_p,
        lc2_g, lc2_b, lc2_m, lc2_v, A, 128);                              // look_conv2
    KA<<<g128, 256, SH3, stream>>>(xh, nullptr, p2c1_p,
        p2c1_g, p2c1_b, p2c1_m, p2c1_v, Bf, 128);                         // p2_conv1
    top_pool_nhwc<<<pool_blocks, 256, 0, stream>>>(A);                    // look1
    KB<<<g128, 256, SH3, stream>>>(Bf, A, p2lc_p,
        nullptr, nullptr, nullptr, nullptr, Df, 128);                     // p2_look
    left_pool_nhwc<<<pool_blocks, 256, 0, stream>>>(Df);                  // pool2

    // merge
    KC<<<g256, 256, SH3, stream>>>(Cf, Df, pconv1_p,
        pbn1_g, pbn1_b, pbn1_m, pbn1_v, E, 256);                          // p_bn1
    KD<<<g256, 256, SH1, stream>>>(xh, nullptr, conv1_p,
        bn1_g, bn1_b, bn1_m, bn1_v, F, 256);                              // bn1
    // out = ReLU(BN(conv3x3(ReLU(p_bn1 + bn1)))) -> f32 NCHW
    KE<<<g256, 256, SH3, stream>>>(E, F, c2_p,
        c2_g, c2_b, c2_m, c2_v, outp, 256);
}
